// SwinTransformerBlock_9646496547095
// MI455X (gfx1250) — compile-verified, matrix operand order checked by reading, not run on hardware
//
#include <hip/hip_runtime.h>
#include <hip/hip_bf16.h>
#include <math.h>

// ---------------------------------------------------------------------------
// Problem constants (from reference)
// ---------------------------------------------------------------------------
#define BQ      2
#define Hdim    128
#define Wdim    128
#define DIM     384
#define NH      12
#define WS      8
#define NWIN    64              // WS*WS tokens per window
#define HD      32              // DIM/NH
#define HIDDEN  1536
#define MTOK    (BQ * Hdim * Wdim)   // 32768 rows
#define QKVN    (3 * DIM)            // 1152

typedef __bf16 v16bf __attribute__((ext_vector_type(16)));
typedef float  v8f   __attribute__((ext_vector_type(8)));
typedef unsigned int v4u __attribute__((ext_vector_type(4)));
typedef int v4i __attribute__((ext_vector_type(4)));

union Frag32B { v4u u[2]; v16bf v; };
union Pack16B { v4u u; __bf16 e[8]; };

// ---------------------------------------------------------------------------
// CDNA5 async global->LDS copies (ASYNCcnt-tracked), if the toolchain has them
// Prototype (probed from hipcc diagnostic): (int4 as1*, int4 as3*, imm, imm)
// ---------------------------------------------------------------------------
#if defined(__has_builtin)
# if __has_builtin(__builtin_amdgcn_global_load_async_to_lds_b128) && \
     __has_builtin(__builtin_amdgcn_s_wait_asynccnt)
#  define USE_ASYNC_LDS 1
# endif
#endif
#ifndef USE_ASYNC_LDS
# define USE_ASYNC_LDS 0
#endif

#if USE_ASYNC_LDS
#define AS1Q __attribute__((address_space(1)))
#define AS3Q __attribute__((address_space(3)))
__device__ __forceinline__ void async_cp128(const __bf16* g, __bf16* l) {
    __builtin_amdgcn_global_load_async_to_lds_b128(
        (AS1Q v4i*)(v4i*)(void*)(g),
        (AS3Q v4i*)(v4i*)(void*)(l),
        0, 0);
}
#define ASYNC_WAIT0() __builtin_amdgcn_s_wait_asynccnt(0)
#endif

// ---------------------------------------------------------------------------
// Load one 16x32 bf16 WMMA operand fragment from LDS.
// ISA 16-bit A layout (16x32): lanes 0-15 hold row M=lane, K {0..7,16..23};
// lanes 16-31 hold row M=lane-16, K {8..15,24..31}. B fragments use the same
// gather with "row" = output column (weights stored N x K row-major).
// ---------------------------------------------------------------------------
__device__ __forceinline__ v16bf ld_frag(const __bf16* base, int stride_e, int kbase) {
    int l   = threadIdx.x & 31;
    int row = l & 15;
    const __bf16* p = base + row * stride_e + kbase + ((l & 16) ? 8 : 0);
    Frag32B f;
    f.u[0] = *reinterpret_cast<const v4u*>(p);
    f.u[1] = *reinterpret_cast<const v4u*>(p + 16);
    return f.v;
}

// ---------------------------------------------------------------------------
// f32 -> bf16 conversion (weights)
// ---------------------------------------------------------------------------
__global__ __launch_bounds__(256) void cvt_bf16_kernel(const float* __restrict__ in,
                                                       __bf16* __restrict__ out, int n) {
    int i = blockIdx.x * 256 + threadIdx.x;
    if (i < n) out[i] = (__bf16)in[i];
}

// ---------------------------------------------------------------------------
// LayerNorm: one wave per row of 384, output bf16
// ---------------------------------------------------------------------------
__global__ __launch_bounds__(256) void ln_kernel(const float* __restrict__ x,
                                                 const float* __restrict__ w,
                                                 const float* __restrict__ bsrc,
                                                 __bf16* __restrict__ out, int rows) {
    int wave = threadIdx.x >> 5, lane = threadIdx.x & 31;
    int row = blockIdx.x * 8 + wave;
    if (row >= rows) return;
    const float* xr = x + (size_t)row * DIM;
    float v[12];
    float s = 0.f;
    #pragma unroll
    for (int i = 0; i < 12; ++i) { v[i] = xr[lane + i * 32]; s += v[i]; }
    #pragma unroll
    for (int off = 16; off; off >>= 1) s += __shfl_xor(s, off, 32);
    float mu = s * (1.f / DIM);
    float q = 0.f;
    #pragma unroll
    for (int i = 0; i < 12; ++i) { float d = v[i] - mu; q += d * d; }
    #pragma unroll
    for (int off = 16; off; off >>= 1) q += __shfl_xor(q, off, 32);
    float rstd = rsqrtf(q * (1.f / DIM) + 1e-5f);
    __bf16* orow = out + (size_t)row * DIM;
    #pragma unroll
    for (int i = 0; i < 12; ++i) {
        int c = lane + i * 32;
        orow[c] = (__bf16)((v[i] - mu) * rstd * w[c] + bsrc[c]);
    }
}

// ---------------------------------------------------------------------------
// Generic bf16 WMMA GEMM: C(M,N) = A(MxK bf16) @ W(NxK bf16)^T + bias
// Block tile 128x128, 8 waves in 4x2, each wave 32x64 (2x4 wmma 16x16x32).
// Double-buffered LDS, async global->LDS staging when available.
// EPI: 0 = bf16 store, 1 = resid + f32 store, 2 = GELU + bf16 store
// ---------------------------------------------------------------------------
template <int EPI>
__global__ __launch_bounds__(256) void gemm_bf16_kernel(const __bf16* __restrict__ A,
                                                        const __bf16* __restrict__ Wt,
                                                        const float* __restrict__ bias,
                                                        const float* __restrict__ resid,
                                                        void* __restrict__ out,
                                                        int M, int N, int K) {
    __shared__ __bf16 sA[2][128 * 32];
    __shared__ __bf16 sB[2][128 * 32];

    int tid  = threadIdx.x;
    int wave = tid >> 5, lane = tid & 31;
    int m0 = blockIdx.x * 128;
    int n0 = blockIdx.y * 128;
    int wm = wave & 3, wn = wave >> 2;

    // staging mapping: 128 rows x 32 cols per tile, 16 bf16 per thread
    int lrow = tid >> 1, lhalf = tid & 1;
    const __bf16* gA = A  + (size_t)(m0 + lrow) * K + lhalf * 16;
    const __bf16* gB = Wt + (size_t)(n0 + lrow) * K + lhalf * 16;
    int loff = lrow * 32 + lhalf * 16;

    v8f acc[2][4] = {};

    auto compute = [&](const __bf16* bufA, const __bf16* bufB) {
        v16bf a0 = ld_frag(bufA + (wm * 32 + 0)  * 32, 32, 0);
        v16bf a1 = ld_frag(bufA + (wm * 32 + 16) * 32, 32, 0);
        #pragma unroll
        for (int tn = 0; tn < 4; ++tn) {
            v16bf b = ld_frag(bufB + (wn * 64 + tn * 16) * 32, 32, 0);
            acc[0][tn] = __builtin_amdgcn_wmma_f32_16x16x32_bf16(false, a0, false, b, (short)0, acc[0][tn], false, false);
            acc[1][tn] = __builtin_amdgcn_wmma_f32_16x16x32_bf16(false, a1, false, b, (short)0, acc[1][tn], false, false);
        }
    };

    int KT = K >> 5;
#if USE_ASYNC_LDS
    // prologue: stage tile 0 into buffer 0
    async_cp128(gA, &sA[0][loff]);
    async_cp128(gA + 8, &sA[0][loff + 8]);
    async_cp128(gB, &sB[0][loff]);
    async_cp128(gB + 8, &sB[0][loff + 8]);
    for (int kt = 0; kt < KT; ++kt) {
        ASYNC_WAIT0();          // own wave's batch kt landed in LDS
        __syncthreads();        // everyone's batch kt visible; prev reads done
        if (kt + 1 < KT) {      // stage next tile into other buffer
            int k0 = (kt + 1) << 5;
            int nb = (kt + 1) & 1;
            async_cp128(gA + k0, &sA[nb][loff]);
            async_cp128(gA + k0 + 8, &sA[nb][loff + 8]);
            async_cp128(gB + k0, &sB[nb][loff]);
            async_cp128(gB + k0 + 8, &sB[nb][loff + 8]);
        }
        compute(sA[kt & 1], sB[kt & 1]);
    }
#else
    for (int kt = 0; kt < KT; ++kt) {
        int k0 = kt << 5;
        __syncthreads();
        {
            const v4u* g = reinterpret_cast<const v4u*>(gA + k0);
            v4u x0 = g[0], x1 = g[1];
            v4u* s = reinterpret_cast<v4u*>(&sA[0][loff]);
            s[0] = x0; s[1] = x1;
            const v4u* g2 = reinterpret_cast<const v4u*>(gB + k0);
            v4u y0 = g2[0], y1 = g2[1];
            v4u* s2 = reinterpret_cast<v4u*>(&sB[0][loff]);
            s2[0] = y0; s2[1] = y1;
            if (k0 + 32 < K) {
                __builtin_prefetch(gA + k0 + 32, 0, 0);
                __builtin_prefetch(gB + k0 + 32, 0, 0);
            }
        }
        __syncthreads();
        compute(sA[0], sB[0]);
    }
#endif

    int Nl = lane & 15;
    int Moff = (lane >> 4) << 3;
    #pragma unroll
    for (int tm = 0; tm < 2; ++tm)
        #pragma unroll
        for (int tn = 0; tn < 4; ++tn) {
            int col = n0 + wn * 64 + tn * 16 + Nl;
            float bv = bias[col];
            #pragma unroll
            for (int vg = 0; vg < 8; ++vg) {
                int row = m0 + wm * 32 + tm * 16 + Moff + vg;
                float v = acc[tm][tn][vg] + bv;
                size_t oi = (size_t)row * N + col;
                if constexpr (EPI == 0) {
                    ((__bf16*)out)[oi] = (__bf16)v;
                } else if constexpr (EPI == 1) {
                    ((float*)out)[oi] = resid[oi] + v;
                } else {
                    float g = 0.5f * v * (1.0f + erff(v * 0.70710678118654752f));
                    ((__bf16*)out)[oi] = (__bf16)g;
                }
            }
        }
}

// ---------------------------------------------------------------------------
// Windowed attention: one block per window (512 blocks, 256 threads).
// Computes haversine distance matrix once, then loops 12 heads:
//   S = scale*q@k^T + sphere-bias  (WMMA), softmax (LDS), out = P@V (WMMA)
// ---------------------------------------------------------------------------
__global__ __launch_bounds__(256) void attn_kernel(const __bf16* __restrict__ qkv,
                                                   const float* __restrict__ uv,
                                                   const float* __restrict__ alpha,
                                                   const float* __restrict__ beta,
                                                   __bf16* __restrict__ out) {
    __shared__ float  sd[NWIN * NWIN];
    __shared__ float  Ss[NWIN * NWIN];
    __shared__ __bf16 probs[NWIN * NWIN];
    __shared__ __bf16 qs[NWIN * HD];
    __shared__ __bf16 ks_[NWIN * HD];
    __shared__ __bf16 vT[HD * NWIN];
    __shared__ float  lon[NWIN], lat[NWIN];

    int w = blockIdx.x;
    int b = w >> 8;          // 256 windows per batch image
    int rem = w & 255;
    int wh = rem >> 4, ww = rem & 15;
    int tid = threadIdx.x;
    int wave = tid >> 5, lane = tid & 31;
    int Nl = lane & 15, Moff = (lane >> 4) << 3;

    auto rowOf = [&](int n) -> size_t {
        return (size_t)b * (Hdim * Wdim) + (size_t)(wh * WS + (n >> 3)) * Wdim + ww * WS + (n & 7);
    };

    if (tid < NWIN) {
        size_t r = rowOf(tid);
        lon[tid] = uv[r * 2 + 0];
        lat[tid] = uv[r * 2 + 1];
    }
    __syncthreads();
    for (int e = tid; e < NWIN * NWIN; e += 256) {
        int n = e >> 6, m = e & 63;
        float dlat = lat[m] - lat[n];
        float dlon = lon[m] - lon[n];
        float s1 = __sinf(dlat * 0.5f);
        float s2 = __sinf(dlon * 0.5f);
        float a = s1 * s1 + __cosf(lat[n]) * __cosf(lat[m]) * s2 * s2;
        a = fminf(fmaxf(a, 0.f), 1.f);
        sd[e] = 2.f * asinf(sqrtf(a));
    }
    // right_u at window token n = N/2 = 32 -> col = ww*8; 1 if col < (W+1)/2
    float rp = (ww * WS < (Wdim + 1) / 2) ? 1.f : 0.f;
    const float scale = 0.17677669529663687f;   // HD^-0.5

    for (int h = 0; h < NH; ++h) {
        __syncthreads();    // protects sd on first iter, LDS reuse after
        {   // stage q, k row-major and v transposed into LDS
            int row = tid >> 2, ch = tid & 3;
            size_t r = rowOf(row);
            const __bf16* base = qkv + r * QKVN + h * HD + ch * 8;
#if USE_ASYNC_LDS
            async_cp128(base,       qs  + row * HD + ch * 8);
            async_cp128(base + DIM, ks_ + row * HD + ch * 8);
#else
            *reinterpret_cast<v4u*>(qs + row * HD + ch * 8)  = *reinterpret_cast<const v4u*>(base);
            *reinterpret_cast<v4u*>(ks_ + row * HD + ch * 8) = *reinterpret_cast<const v4u*>(base + DIM);
#endif
            Pack16B vv;
            vv.u = *reinterpret_cast<const v4u*>(base + 2 * DIM);
            #pragma unroll
            for (int i = 0; i < 8; ++i) vT[(ch * 8 + i) * NWIN + row] = vv.e[i];
#if USE_ASYNC_LDS
            ASYNC_WAIT0();
#endif
        }
        __syncthreads();
        {   // S = scale * q @ k^T + bias ; 16 tiles over 8 waves (2 each)
            int mt = wave & 3, ntB = (wave >> 2) * 2;
            v16bf aq = ld_frag(qs + mt * 16 * HD, HD, 0);
            #pragma unroll
            for (int t = 0; t < 2; ++t) {
                int nt = ntB + t;
                v16bf bk = ld_frag(ks_ + nt * 16 * HD, HD, 0);
                v8f c = {};
                c = __builtin_amdgcn_wmma_f32_16x16x32_bf16(false, aq, false, bk, (short)0, c, false, false);
                #pragma unroll
                for (int vg = 0; vg < 8; ++vg) {
                    int n = mt * 16 + Moff + vg;      // query index
                    int m = nt * 16 + Nl;             // key index
                    int im = m >> 3, jm = m & 7, in_ = n >> 3, jn = n & 7;
                    int idx0 = (im - in_ + 7) * 15 + (jm - jn + 7);
                    int idx1 = (in_ - im + 7) * 15 + (jn - jm + 7);
                    float t0 = alpha[idx0 * NH + h] * beta[idx0 * NH + h];
                    float t1 = alpha[idx1 * NH + h] * beta[idx1 * NH + h];
                    float bias = sd[m * NWIN + n] * (rp * t0 + (1.f - rp) * t1);
                    Ss[n * NWIN + m] = c[vg] * scale + bias;
                }
            }
        }
        __syncthreads();
        if (tid < NWIN) {   // softmax over m
            int n = tid;
            float mx = -3.0e38f;
            for (int m = 0; m < NWIN; ++m) mx = fmaxf(mx, Ss[n * NWIN + m]);
            float sum = 0.f;
            for (int m = 0; m < NWIN; ++m) sum += __expf(Ss[n * NWIN + m] - mx);
            float inv = 1.f / sum;
            for (int m = 0; m < NWIN; ++m)
                probs[n * NWIN + m] = (__bf16)(__expf(Ss[n * NWIN + m] - mx) * inv);
        }
        __syncthreads();
        {   // out = P @ V ; 8 tiles (4x2) over 8 waves, K = 64 in two steps
            int mt = wave & 3, dt = wave >> 2;
            v8f c = {};
            #pragma unroll
            for (int kb = 0; kb < NWIN; kb += 32) {
                v16bf ap = ld_frag(probs + mt * 16 * NWIN, NWIN, kb);
                v16bf bv = ld_frag(vT + dt * 16 * NWIN, NWIN, kb);
                c = __builtin_amdgcn_wmma_f32_16x16x32_bf16(false, ap, false, bv, (short)0, c, false, false);
            }
            #pragma unroll
            for (int vg = 0; vg < 8; ++vg) {
                int n = mt * 16 + Moff + vg;
                int d = dt * 16 + Nl;
                size_t r = rowOf(n);
                out[r * DIM + h * HD + d] = (__bf16)c[vg];
            }
        }
    }
}

// ---------------------------------------------------------------------------
// Host launch
// ---------------------------------------------------------------------------
extern "C" void kernel_launch(void* const* d_in, const int* in_sizes, int n_in,
                              void* d_out, int out_size, void* d_ws, size_t ws_size,
                              hipStream_t stream) {
    (void)in_sizes; (void)n_in; (void)out_size; (void)ws_size;

    const float* x        = (const float*)d_in[0];
    const float* uv       = (const float*)d_in[1];
    const float* norm1_w  = (const float*)d_in[2];
    const float* norm1_b  = (const float*)d_in[3];
    const float* qkv_w    = (const float*)d_in[4];
    const float* qkv_b    = (const float*)d_in[5];
    const float* alpha_t  = (const float*)d_in[6];
    const float* beta_t   = (const float*)d_in[7];
    const float* proj_w   = (const float*)d_in[8];
    const float* proj_b   = (const float*)d_in[9];
    const float* norm2_w  = (const float*)d_in[10];
    const float* norm2_b  = (const float*)d_in[11];
    const float* fc1_w    = (const float*)d_in[12];
    const float* fc1_b    = (const float*)d_in[13];
    const float* fc2_w    = (const float*)d_in[14];
    const float* fc2_b    = (const float*)d_in[15];
    float* outp = (float*)d_out;

    char* ws = (char*)d_ws;
    // Workspace layout (bytes), regions reused across phases:
    const size_t R1 = 0;                         // qkv bf16 (75.5MB) then FC1 hidden bf16 (100.7MB)
    const size_t R2 = R1 + (size_t)MTOK * HIDDEN * 2;   // xn / xn2 bf16 (25.2MB)
    const size_t R3 = R2 + (size_t)MTOK * DIM * 2;      // attn_out bf16 (25.2MB)
    const size_t R4 = R3 + (size_t)MTOK * DIM * 2;      // x2 f32 (50.3MB)
    const size_t R5 = R4 + (size_t)MTOK * DIM * 4;      // converted weights
    __bf16* qkvb   = (__bf16*)(ws + R1);
    __bf16* hbuf   = (__bf16*)(ws + R1);
    __bf16* xn     = (__bf16*)(ws + R2);
    __bf16* xn2    = (__bf16*)(ws + R2);
    __bf16* aout   = (__bf16*)(ws + R3);
    float*  x2     = (float*)(ws + R4);
    __bf16* wq     = (__bf16*)(ws + R5);
    __bf16* wp     = wq + (size_t)QKVN * DIM;
    __bf16* w1     = wp + (size_t)DIM * DIM;
    __bf16* w2     = w1 + (size_t)HIDDEN * DIM;

    // 1) weight conversion
    {
        int n;
        n = QKVN * DIM;   cvt_bf16_kernel<<<(n + 255) / 256, 256, 0, stream>>>(qkv_w, wq, n);
        n = DIM * DIM;    cvt_bf16_kernel<<<(n + 255) / 256, 256, 0, stream>>>(proj_w, wp, n);
        n = HIDDEN * DIM; cvt_bf16_kernel<<<(n + 255) / 256, 256, 0, stream>>>(fc1_w, w1, n);
        n = DIM * HIDDEN; cvt_bf16_kernel<<<(n + 255) / 256, 256, 0, stream>>>(fc2_w, w2, n);
    }
    // 2) LN1
    ln_kernel<<<MTOK / 8, 256, 0, stream>>>(x, norm1_w, norm1_b, xn, MTOK);
    // 3) QKV GEMM: 32768 x 1152 x 384
    gemm_bf16_kernel<0><<<dim3(MTOK / 128, QKVN / 128), 256, 0, stream>>>(
        xn, wq, qkv_b, nullptr, (void*)qkvb, MTOK, QKVN, DIM);
    // 4) windowed attention
    attn_kernel<<<512, 256, 0, stream>>>(qkvb, uv, alpha_t, beta_t, aout);
    // 5) proj GEMM + residual -> x2 (f32)
    gemm_bf16_kernel<1><<<dim3(MTOK / 128, DIM / 128), 256, 0, stream>>>(
        aout, wp, proj_b, x, (void*)x2, MTOK, DIM, DIM);
    // 6) LN2
    ln_kernel<<<MTOK / 8, 256, 0, stream>>>(x2, norm2_w, norm2_b, xn2, MTOK);
    // 7) FC1 + GELU: 32768 x 1536 x 384
    gemm_bf16_kernel<2><<<dim3(MTOK / 128, HIDDEN / 128), 256, 0, stream>>>(
        xn2, w1, fc1_b, nullptr, (void*)hbuf, MTOK, HIDDEN, DIM);
    // 8) FC2 + residual: 32768 x 384 x 1536 -> d_out (f32)
    gemm_bf16_kernel<1><<<dim3(MTOK / 128, DIM / 128), 256, 0, stream>>>(
        hbuf, w2, fc2_b, x2, (void*)outp, MTOK, DIM, HIDDEN);
}
